// MultiHeadLatentAttention_2508260900992
// MI455X (gfx1250) — matrix operand order repaired and checked by reading, not compiled, not run on hardware
//
#include <hip/hip_runtime.h>

typedef unsigned short u16;
typedef unsigned int   u32;
typedef unsigned long long u64;
typedef __attribute__((ext_vector_type(16))) __bf16 v16bf;
typedef __attribute__((ext_vector_type(16))) u16    v16u;
typedef __attribute__((ext_vector_type(8)))  float  v8f;
typedef __attribute__((ext_vector_type(4)))  u32    v4u;
typedef __attribute__((ext_vector_type(8)))  int    v8i;
typedef __attribute__((ext_vector_type(4)))  int    v4i;

#define B_SZ   2
#define S_LEN  2048
#define BS     4096      // B*S
#define E_DIM  2048
#define H_Q    16
#define HKV    4
#define QLR    1024
#define KVLR   512
#define DN     128
#define DR     64
#define DV     128
#define DQK    192
#define EPS_F  1.1920929e-07f
#define NEG_INF (-__builtin_inff())

// ---------------- helpers ----------------
__device__ __forceinline__ u16 f2bf(float f) {          // RNE f32 -> bf16
  u32 u = __builtin_bit_cast(u32, f);
  u32 r = u + 0x7FFFu + ((u >> 16) & 1u);
  return (u16)(r >> 16);
}
__device__ __forceinline__ float bf2f(u16 h) {
  u32 u = ((u32)h) << 16;
  return __builtin_bit_cast(float, u);
}
// CDNA5 WMMA bf16: D(16x16,f32) = A(16x32) * B(32x16) + C
__device__ __forceinline__ v8f wmma_bf16(v16u a, v16u b, v8f c) {
  return __builtin_amdgcn_wmma_f32_16x16x32_bf16(
      false, __builtin_bit_cast(v16bf, a),
      false, __builtin_bit_cast(v16bf, b),
      (short)0, c, false, false);
}
// A-matrix (16x32, 16-bit) per-lane K start for VGPR-pair i (ISA 7.12.2)
__device__ __forceinline__ int a_kbase(int i, int lh) {
  int base = (i < 4) ? (2 * i) : (16 + 2 * (i - 4));
  return base + (lh ? 8 : 0);
}
__device__ __forceinline__ float redmax16(float v) {
#pragma unroll
  for (int off = 1; off < 16; off <<= 1) v = fmaxf(v, __shfl_xor(v, off, 32));
  return v;
}
__device__ __forceinline__ float redsum16(float v) {
#pragma unroll
  for (int off = 1; off < 16; off <<= 1) v += __shfl_xor(v, off, 32);
  return v;
}

// ---- Tensor Data Mover: 2D tile load (bf16 elems) with LDS row padding ----
// D# layout per CDNA5 ISA.
// This toolchain's builtin (clang-23 form):
//   (uint32x4 g0, int32x8 g1, int32x4 g2, int32x4 g3, int32x8, i32 cpol)
__device__ __forceinline__ void tdm_load_2d(
    u32 lds_off, const void* gptr, u32 tile0, u32 tile1, u64 stride0_elems,
    u32 pad_interval_code, u32 pad_amount_code) {
  u64 ga = (u64)gptr;
  v4u g0;
  g0[0] = 1u;                                   // count=1, user mode
  g0[1] = lds_off;                              // lds_addr (bytes)
  g0[2] = (u32)ga;                              // global_addr[31:0]
  g0[3] = (u32)((ga >> 32) & 0x1FFFFFFu) | (2u << 30);  // addr[56:32] | type=2
  u32 td0 = tile0;                              // tensor_dim0 == tile width
  u32 td1 = 1u << 20;                           // tensor_dim1: large (no OOB)
  v8i g1;
  g1[0] = (int)((1u << 16)                      // data_size = 1 (2 bytes)
               | (1u << 20)                     // pad_enable
               | (pad_interval_code << 22)
               | (pad_amount_code << 25));
  g1[1] = (int)((td0 & 0xFFFFu) << 16);         // tensor_dim0[15:0] (bits 63:48)
  g1[2] = (int)((td0 >> 16) | ((td1 & 0xFFFFu) << 16));
  g1[3] = (int)((td1 >> 16) | (tile0 << 16));   // tile_dim0
  g1[4] = (int)(tile1);                         // tile_dim1 (tile_dim2=0)
  g1[5] = (int)(u32)stride0_elems;              // tensor_dim0_stride[31:0]
  g1[6] = (int)(u32)((stride0_elems >> 32) & 0xFFFFu);
  g1[7] = 0;
  v4i gz4 = (v4i){0, 0, 0, 0};
  v8i gz8 = (v8i){0, 0, 0, 0, 0, 0, 0, 0};
  __builtin_amdgcn_tensor_load_to_lds(g0, g1, gz4, gz4, gz8, 0);
}

// ---------------- fp32 -> bf16 cast ----------------
__global__ __launch_bounds__(256) void cast_bf16_kernel(
    const float* __restrict__ in, u16* __restrict__ out, long n) {
  long i = (long)blockIdx.x * 256 + threadIdx.x;
  if (i < n) out[i] = f2bf(in[i]);
}

// ---------------- generic bf16 GEMM:  C[M,N] = A[M,K] * W[N,K]^T --------------
// block = 128 threads = 4 waves; block tile 128x64; wave tile 32x64; K step 32.
template <bool OUT_BF16>
__global__ __launch_bounds__(128) void gemm_bf16_kernel(
    const u16* __restrict__ A, const u16* __restrict__ W, void* __restrict__ Cout,
    int K, int lda, int ldb, int ldc) {
  __shared__ __align__(16) u16 Xs[128][32];
  __shared__ __align__(16) u16 Ws[64][32];
  const int tid = threadIdx.x, lane = tid & 31, wv = tid >> 5;
  const int lh = lane >> 4, mr = lane & 15;
  const long m_blk = (long)blockIdx.y * 128;
  const long n_blk = (long)blockIdx.x * 64;

  v8f acc[2][4];
#pragma unroll
  for (int mt = 0; mt < 2; ++mt)
#pragma unroll
    for (int t = 0; t < 4; ++t)
      acc[mt][t] = (v8f){0.f,0.f,0.f,0.f,0.f,0.f,0.f,0.f};

  for (int k0 = 0; k0 < K; k0 += 32) {
    __syncthreads();
#pragma unroll
    for (int it = 0; it < 4; ++it) {           // X: 128 rows * 4 chunks
      int idx = tid + it * 128;
      int row = idx >> 2, c4 = idx & 3;
      *((uint4*)&Xs[row][c4 * 8]) =
          *(const uint4*)&A[(m_blk + row) * (long)lda + k0 + c4 * 8];
    }
#pragma unroll
    for (int it = 0; it < 2; ++it) {           // W: 64 rows * 4 chunks
      int idx = tid + it * 128;
      int row = idx >> 2, c4 = idx & 3;
      *((uint4*)&Ws[row][c4 * 8]) =
          *(const uint4*)&W[(n_blk + row) * (long)ldb + k0 + c4 * 8];
    }
    __syncthreads();
    v16u a[2];
#pragma unroll
    for (int mt = 0; mt < 2; ++mt)
#pragma unroll
      for (int i = 0; i < 8; ++i) {
        u32 u = *(const u32*)&Xs[wv * 32 + mt * 16 + mr][a_kbase(i, lh)];
        a[mt][2 * i] = (u16)u; a[mt][2 * i + 1] = (u16)(u >> 16);
      }
#pragma unroll
    for (int t = 0; t < 4; ++t) {
      v16u bfr;
#pragma unroll
      for (int i = 0; i < 8; ++i) {
        u32 u = *(const u32*)&Ws[t * 16 + mr][a_kbase(i, lh)];
        bfr[2 * i] = (u16)u; bfr[2 * i + 1] = (u16)(u >> 16);
      }
#pragma unroll
      for (int mt = 0; mt < 2; ++mt)
        acc[mt][t] = wmma_bf16(a[mt], bfr, acc[mt][t]);
    }
  }
#pragma unroll
  for (int mt = 0; mt < 2; ++mt)
#pragma unroll
    for (int t = 0; t < 4; ++t)
#pragma unroll
      for (int r = 0; r < 8; ++r) {
        long row = m_blk + wv * 32 + mt * 16 + r + 8 * lh;
        long col = n_blk + t * 16 + mr;
        if (OUT_BF16) ((u16*)Cout)[row * ldc + col] = f2bf(acc[mt][t][r]);
        else          ((float*)Cout)[row * ldc + col] = acc[mt][t][r];
      }
}

// ---------------- row RMS norm (f32 in, bf16 out) ----------------
__global__ __launch_bounds__(256) void rms_bf16_kernel(
    const float* __restrict__ in, u16* __restrict__ out,
    int L, int ldin, int ldout) {
  const long row = blockIdx.x;
  const int tid = threadIdx.x, lane = tid & 31, wv = tid >> 5;
  float ss = 0.f;
  for (int c = tid; c < L; c += 256) { float v = in[row * ldin + c]; ss += v * v; }
#pragma unroll
  for (int off = 1; off < 32; off <<= 1) ss += __shfl_xor(ss, off, 32);
  __shared__ float red[8];
  if (lane == 0) red[wv] = ss;
  __syncthreads();
  float tot = 0.f;
#pragma unroll
  for (int w = 0; w < 8; ++w) tot += red[w];
  float sc = rsqrtf(tot / (float)L + EPS_F);
  for (int c = tid; c < L; c += 256)
    out[row * ldout + c] = f2bf(in[row * ldin + c] * sc);
}

// ---------------- in-place RoPE on q (bf16, layout (BS, H, 192)) -------------
__global__ __launch_bounds__(256) void rope_q_kernel(
    u16* __restrict__ q, const float* __restrict__ fc, const float* __restrict__ fs) {
  long i = (long)blockIdx.x * 256 + threadIdx.x;   // over BS*H*32 pairs
  if (i >= (long)BS * H_Q * 32) return;
  int p = (int)(i & 31);
  long rem = i >> 5;
  int h = (int)(rem % H_Q);
  long bs = rem / H_Q;
  int s = (int)(bs % S_LEN);
  float c = fc[(long)s * 32 + p], sn = fs[(long)s * 32 + p];
  long base = (bs * H_Q + h) * DQK + DN + 2 * p;
  float x0 = bf2f(q[base]), x1 = bf2f(q[base + 1]);
  q[base]     = f2bf(x0 * c - x1 * sn);
  q[base + 1] = f2bf(x0 * sn + x1 * c);
}

// ---------------- RoPE on k_rope (f32 kv_a[:,512:576] -> bf16 (BS,64)) -------
__global__ __launch_bounds__(256) void rope_k_kernel(
    const float* __restrict__ kva, u16* __restrict__ kr,
    const float* __restrict__ fc, const float* __restrict__ fs) {
  long i = (long)blockIdx.x * 256 + threadIdx.x;   // over BS*32 pairs
  if (i >= (long)BS * 32) return;
  int p = (int)(i & 31);
  long bs = i >> 5;
  int s = (int)(bs % S_LEN);
  float c = fc[(long)s * 32 + p], sn = fs[(long)s * 32 + p];
  float x0 = kva[bs * 576 + KVLR + 2 * p];
  float x1 = kva[bs * 576 + KVLR + 2 * p + 1];
  kr[bs * DR + 2 * p]     = f2bf(x0 * c - x1 * sn);
  kr[bs * DR + 2 * p + 1] = f2bf(x0 * sn + x1 * c);
}

// ---------------- flash attention core ----------------
// grid = (S/16, B*HKV); block = 128 threads = 4 waves (one query head each).
// q  : (BS, 16, 192) bf16 (post-RoPE)
// kv : (BS, 4, 256)  bf16  [0:128)=k_nope, [128:256)=v
// kr : (BS, 64)      bf16  (roped k, shared across KV heads)
// o  : (BS, 16, 128) bf16
__global__ __launch_bounds__(128) void mla_attn_kernel(
    const u16* __restrict__ q, const u16* __restrict__ kv,
    const u16* __restrict__ kr, u16* __restrict__ o) {
  __shared__ __align__(16) u16 Klds[32][192];   // 32 keys x DQK (TDM-filled)
  __shared__ __align__(16) u16 Vt[128][32];     // transposed: [v-dim][key]
  __shared__ __align__(16) u16 Plds[4][16][32]; // per-wave P staging
  const int tid = threadIdx.x, lane = tid & 31, wv = tid >> 5;
  const int lh = lane >> 4, mr = lane & 15;
  const int bb = blockIdx.y >> 2, hk = blockIdx.y & 3;
  const int h = hk * 4 + wv;               // query head for this wave
  const int q0 = blockIdx.x * 16;
  const long bs0 = (long)bb * S_LEN;
  const u32 ldsK = (u32)(unsigned long long)(&Klds[0][0]);  // LDS byte offset

  // Q fragments (16 rows x 192) as six 16x32 A-fragments
  v16u aq[6];
#pragma unroll
  for (int kc = 0; kc < 6; ++kc)
#pragma unroll
    for (int i = 0; i < 8; ++i) {
      long idx = ((bs0 + q0 + mr) * H_Q + h) * (long)DQK + kc * 32 + a_kbase(i, lh);
      u32 u = *(const u32*)&q[idx];
      aq[kc][2 * i] = (u16)u; aq[kc][2 * i + 1] = (u16)(u >> 16);
    }

  float m_i[8], l_i[8];
  v8f oacc[8];
#pragma unroll
  for (int r = 0; r < 8; ++r) { m_i[r] = NEG_INF; l_i[r] = 0.f; }
#pragma unroll
  for (int t = 0; t < 8; ++t) oacc[t] = (v8f){0.f,0.f,0.f,0.f,0.f,0.f,0.f,0.f};

  const int nblk = (q0 + 15) / 32 + 1;     // key blocks of 32 with causal cut
  for (int blk = 0; blk < nblk; ++blk) {
    const int k0b = blk * 32;
    __syncthreads();
    // --- K tile via Tensor Data Mover (wave 0 only; per-wave instruction) ---
    if (wv == 0) {
      // k_nope: 32 rows x 128 elems, row stride 1024 elems;
      // LDS rows are 384B: pad every 256B (interval code 5 = 64 dw) by
      // 128B (amount code 31 = 32 dw).
      tdm_load_2d(ldsK, &kv[((bs0 + k0b) * HKV + hk) * 256],
                  128, 32, 1024, 5, 31);
      // k_rope: 32 rows x 64 elems, row stride 64 elems, dest Klds[.][128];
      // pad every 128B (code 4 = 32 dw) by 256B (code 63 = 64 dw).
      tdm_load_2d(ldsK + 256, &kr[(bs0 + k0b) * DR],
                  64, 32, 64, 4, 63);
      __builtin_amdgcn_s_wait_tensorcnt(0);
    }
    // --- V tile, transposed, manual (TDM cannot transpose) ---
    for (int idx = tid; idx < 2048; idx += 128) {
      int key = idx & 31, vd2 = idx >> 5;
      u32 u = *(const u32*)&kv[((bs0 + k0b + key) * HKV + hk) * 256 + 128 + vd2 * 2];
      Vt[vd2 * 2][key]     = (u16)u;
      Vt[vd2 * 2 + 1][key] = (u16)(u >> 16);
    }
    if (blk + 1 < nblk)   // global_prefetch_b8 next K block
      __builtin_prefetch(&kv[((bs0 + k0b + 32) * HKV + hk) * 256], 0, 0);
    __syncthreads();

    // S = Q * K^T : two 16x16 f32 tiles (keys 0..15 and 16..31)
    v8f s0 = (v8f){0.f,0.f,0.f,0.f,0.f,0.f,0.f,0.f};
    v8f s1 = (v8f){0.f,0.f,0.f,0.f,0.f,0.f,0.f,0.f};
#pragma unroll
    for (int kc = 0; kc < 6; ++kc) {
      v16u b0, b1;
#pragma unroll
      for (int i = 0; i < 8; ++i) {
        int kk = kc * 32 + a_kbase(i, lh);
        u32 u0 = *(const u32*)&Klds[mr][kk];
        u32 u1 = *(const u32*)&Klds[16 + mr][kk];
        b0[2 * i] = (u16)u0; b0[2 * i + 1] = (u16)(u0 >> 16);
        b1[2 * i] = (u16)u1; b1[2 * i + 1] = (u16)(u1 >> 16);
      }
      s0 = wmma_bf16(aq[kc], b0, s0);
      s1 = wmma_bf16(aq[kc], b1, s1);
    }

    // online softmax (row stats live per (lane-half, r) matching C layout)
    const float scale = 0.07216878364870323f;  // 1/sqrt(192)
    float p0[8], p1[8];
#pragma unroll
    for (int r = 0; r < 8; ++r) {
      int qrow = q0 + r + 8 * lh;
      float x0 = (k0b + mr      <= qrow) ? s0[r] * scale : NEG_INF;
      float x1 = (k0b + 16 + mr <= qrow) ? s1[r] * scale : NEG_INF;
      float mnew = fmaxf(m_i[r], redmax16(fmaxf(x0, x1)));
      float alpha = __expf(m_i[r] - mnew);
      p0[r] = __expf(x0 - mnew);
      p1[r] = __expf(x1 - mnew);
      l_i[r] = l_i[r] * alpha + redsum16(p0[r] + p1[r]);
      m_i[r] = mnew;
#pragma unroll
      for (int t = 0; t < 8; ++t) oacc[t][r] *= alpha;
    }

    // P: C-layout -> A-layout via per-wave LDS
#pragma unroll
    for (int r = 0; r < 8; ++r) {
      int row = r + 8 * lh;
      Plds[wv][row][mr]      = f2bf(p0[r]);
      Plds[wv][row][16 + mr] = f2bf(p1[r]);
    }
    __syncthreads();
    v16u ap;
#pragma unroll
    for (int i = 0; i < 8; ++i) {
      u32 u = *(const u32*)&Plds[wv][mr][a_kbase(i, lh)];
      ap[2 * i] = (u16)u; ap[2 * i + 1] = (u16)(u >> 16);
    }
    // O += P * V  (eight 16x16 tiles over DV=128)
#pragma unroll
    for (int t = 0; t < 8; ++t) {
      v16u bv;
#pragma unroll
      for (int i = 0; i < 8; ++i) {
        u32 u = *(const u32*)&Vt[t * 16 + mr][a_kbase(i, lh)];
        bv[2 * i] = (u16)u; bv[2 * i + 1] = (u16)(u >> 16);
      }
      oacc[t] = wmma_bf16(ap, bv, oacc[t]);
    }
  }

  // epilogue: normalize and store (BS, H, 128) bf16
#pragma unroll
  for (int r = 0; r < 8; ++r) {
    float inv = 1.f / l_i[r];
    long row = bs0 + q0 + r + 8 * lh;
#pragma unroll
    for (int t = 0; t < 8; ++t)
      o[(row * H_Q + h) * (long)DV + t * 16 + mr] = f2bf(oacc[t][r] * inv);
  }
}

// ---------------- host-side launch ----------------
static inline dim3 gemm_grid(int N) { return dim3(N / 64, BS / 128); }

extern "C" void kernel_launch(void* const* d_in, const int* in_sizes, int n_in,
                              void* d_out, int out_size, void* d_ws, size_t ws_size,
                              hipStream_t stream) {
  (void)in_sizes; (void)n_in; (void)out_size; (void)ws_size;
  const float* x    = (const float*)d_in[0];   // (BS, 2048)
  const float* fc   = (const float*)d_in[1];   // (S, 32)
  const float* fs   = (const float*)d_in[2];   // (S, 32)
  const float* Wqa  = (const float*)d_in[3];   // (1024, 2048)
  const float* Wqb  = (const float*)d_in[4];   // (3072, 1024)
  const float* Wkva = (const float*)d_in[5];   // (576, 2048)
  const float* Wkvb = (const float*)d_in[6];   // (1024, 512)
  const float* Wo   = (const float*)d_in[7];   // (2048, 2048)

  size_t off = 0;
  auto alloc = [&](size_t bytes) -> char* {
    char* p = (char*)d_ws + off;
    off += (bytes + 255) & ~(size_t)255;
    return p;
  };
  u16*   xb    = (u16*)  alloc((size_t)BS * E_DIM * 2);
  u16*   wqab  = (u16*)  alloc((size_t)QLR * E_DIM * 2);
  u16*   wqbb  = (u16*)  alloc((size_t)3072 * QLR * 2);
  u16*   wkvab = (u16*)  alloc((size_t)576 * E_DIM * 2);
  u16*   wkvbb = (u16*)  alloc((size_t)1024 * KVLR * 2);
  u16*   wob   = (u16*)  alloc((size_t)E_DIM * E_DIM * 2);
  float* t1    = (float*)alloc((size_t)BS * QLR * 4);
  u16*   qlat  = (u16*)  alloc((size_t)BS * QLR * 2);
  u16*   qraw  = (u16*)  alloc((size_t)BS * 3072 * 2);   // becomes roped q
  float* kva   = (float*)alloc((size_t)BS * 576 * 4);
  u16*   kvlat = (u16*)  alloc((size_t)BS * KVLR * 2);
  u16*   kvout = (u16*)  alloc((size_t)BS * 1024 * 2);   // k_nope || v
  u16*   krope = (u16*)  alloc((size_t)BS * DR * 2);
  u16*   ob    = (u16*)  alloc((size_t)BS * 2048 * 2);

  auto cast = [&](const float* src, u16* dst, long n) {
    cast_bf16_kernel<<<dim3((unsigned)((n + 255) / 256)), 256, 0, stream>>>(src, dst, n);
  };
  cast(x,    xb,    (long)BS * E_DIM);
  cast(Wqa,  wqab,  (long)QLR * E_DIM);
  cast(Wqb,  wqbb,  (long)3072 * QLR);
  cast(Wkva, wkvab, (long)576 * E_DIM);
  cast(Wkvb, wkvbb, (long)1024 * KVLR);
  cast(Wo,   wob,   (long)E_DIM * E_DIM);

  // t1 = x * Wqa^T  (f32)
  gemm_bf16_kernel<false><<<gemm_grid(QLR), 128, 0, stream>>>(
      xb, wqab, t1, E_DIM, E_DIM, E_DIM, QLR);
  // qlat = rms(t1) (bf16)
  rms_bf16_kernel<<<BS, 256, 0, stream>>>(t1, qlat, QLR, QLR, QLR);
  // qraw = qlat * Wqb^T (bf16)
  gemm_bf16_kernel<true><<<gemm_grid(3072), 128, 0, stream>>>(
      qlat, wqbb, qraw, QLR, QLR, QLR, 3072);
  // kva = x * Wkva^T (f32)
  gemm_bf16_kernel<false><<<gemm_grid(576), 128, 0, stream>>>(
      xb, wkvab, kva, E_DIM, E_DIM, E_DIM, 576);
  // kvlat = rms(kva[:, :512]) (bf16)
  rms_bf16_kernel<<<BS, 256, 0, stream>>>(kva, kvlat, KVLR, 576, KVLR);
  // kvout = kvlat * Wkvb^T (bf16) -> (BS, 4, 256)
  gemm_bf16_kernel<true><<<gemm_grid(1024), 128, 0, stream>>>(
      kvlat, wkvbb, kvout, KVLR, KVLR, KVLR, 1024);
  // RoPE
  rope_q_kernel<<<dim3((unsigned)(((long)BS * H_Q * 32 + 255) / 256)), 256, 0, stream>>>(
      qraw, fc, fs);
  rope_k_kernel<<<dim3((unsigned)(((long)BS * 32 + 255) / 256)), 256, 0, stream>>>(
      kva, krope, fc, fs);
  // attention: grid (S/16, B*HKV)
  mla_attn_kernel<<<dim3(S_LEN / 16, B_SZ * HKV), 128, 0, stream>>>(
      qraw, kvout, krope, ob);
  // out = o * Wo^T (f32)
  gemm_bf16_kernel<false><<<gemm_grid(E_DIM), 128, 0, stream>>>(
      ob, wob, (float*)d_out, E_DIM, E_DIM, E_DIM, E_DIM);
}
